// GATv2_4131758538795
// MI455X (gfx1250) — matrix operand order repaired and checked by reading, not compiled, not run on hardware
//
#include <hip/hip_runtime.h>

typedef __attribute__((ext_vector_type(16))) _Float16 v16h;
typedef __attribute__((ext_vector_type(8)))  float    v8f;
typedef __attribute__((ext_vector_type(4)))  unsigned v4u;
typedef __attribute__((ext_vector_type(8)))  int      v8i;
typedef __attribute__((ext_vector_type(4)))  int      v4i;

#define N_NODES 50000   // divisible by 16 -> no row guard needed in node GEMM
#define F_IN    128
#define E_EDGES 800000
#define FE_IN   64
#define DMODEL  128
#define NHEADS  8
#define DHEAD   16
#define NKS_W   (F_IN / 32)    // 4 k-steps for node GEMM
#define NKS_WE  (FE_IN / 32)   // 2 k-steps for edge GEMM

// ---------- helpers ----------

__device__ __forceinline__ float mishf(float x) {
    float sp = log1pf(expf(x));      // softplus (inf-safe: tanh(inf)=1)
    return x * tanhf(sp);
}

// Order-preserving encode of f32 into u32 so unsigned atomicMax == float max
__device__ __forceinline__ unsigned enc_f32(float x) {
    unsigned u = __float_as_uint(x);
    return (u & 0x80000000u) ? ~u : (u | 0x80000000u);
}
__device__ __forceinline__ float dec_f32(unsigned e) {
    unsigned u = (e & 0x80000000u) ? (e & 0x7fffffffu) : ~e;
    return __uint_as_float(u);
}

// Build A fragment (16x32 f16, contiguous k order) from a row-major *f16* row in LDS.
// Lane half=0: K = k0+0..15 ; half=1: K = k0+16..31.
// Two contiguous 8-half runs -> 2x ds_load_b128, no conversion in the loop.
__device__ __forceinline__ v16h make_a_frag_h(const _Float16* __restrict__ row,
                                              int k0, int lane) {
    const int kb = (lane >> 4) ? 16 : 0;
    const _Float16* base = row + k0 + kb;
    v16h a;
#pragma unroll
    for (int i = 0; i < 8; ++i) a[i] = base[i];
#pragma unroll
    for (int i = 0; i < 8; ++i) a[8 + i] = base[8 + i];
    return a;
}

// ---------- kernels ----------

__global__ void k_init(float* __restrict__ out, float* __restrict__ nsum,
                       unsigned* __restrict__ nmax) {
    int i = blockIdx.x * blockDim.x + threadIdx.x;
    if (i < N_NODES * DMODEL) out[i] = 0.0f;
    if (i < N_NODES * NHEADS) {
        nsum[i] = 0.0f;
        nmax[i] = enc_f32(-3.4e38f);
    }
}

// Convert weights to f16 *pre-swizzled into WMMA B-fragment order*:
// fragment f = (col_tile*nks + kstep); element j = ((f*32 + lane)*16 + i)
// holds W[k, n] with n = col_tile*16 + (lane&15), k = kstep*32 + (lane>>4)*16 + i.
// Each lane's 16 halves are contiguous (32 B) -> 2x global_load_b128 in the GEMM.
__global__ void k_convert(const float* __restrict__ Wk, const float* __restrict__ Wek,
                          _Float16* __restrict__ Wswz, _Float16* __restrict__ Weswz) {
    int j = blockIdx.x * blockDim.x + threadIdx.x;
    if (j < F_IN * DMODEL) {              // 8 col-tiles * 4 ksteps * 32 lanes * 16
        int i = j & 15, lane = (j >> 4) & 31, f = j >> 9;
        int ks = f & (NKS_W - 1), ct = f / NKS_W;
        int n = ct * 16 + (lane & 15);
        int k = ks * 32 + ((lane >> 4) ? 16 : 0) + i;
        Wswz[j] = (_Float16)Wk[k * DMODEL + n];
    }
    if (j < FE_IN * DMODEL) {             // 8 col-tiles * 2 ksteps * 32 lanes * 16
        int i = j & 15, lane = (j >> 4) & 31, f = j >> 9;
        int ks = f & (NKS_WE - 1), ct = f / NKS_WE;
        int n = ct * 16 + (lane & 15);
        int k = ks * 32 + ((lane >> 4) ? 16 : 0) + i;
        Weswz[j] = (_Float16)Wek[k * DMODEL + n];
    }
}

// proj = node_features @ W + b ; 16-row tile staged once in LDS (as f16),
// wave w -> column tile w.
__global__ __launch_bounds__(256) void k_node_proj(const float* __restrict__ nf,
                                                   const _Float16* __restrict__ Wswz,
                                                   const float* __restrict__ Wb,
                                                   float* __restrict__ proj) {
    __shared__ __align__(16) _Float16 sN[16 * F_IN];   // 4 KB (f16)
    const int t = threadIdx.x;
    const int row0 = blockIdx.x * 16;
    {
        // 2048 f32 -> 2048 f16 ; 8 elements per thread, vector loads + 16B LDS store
        const float4* src = (const float4*)(nf + (size_t)row0 * F_IN);
        float4 x = src[t * 2], y = src[t * 2 + 1];
        _Float16* d = &sN[t * 8];
        d[0] = (_Float16)x.x; d[1] = (_Float16)x.y;
        d[2] = (_Float16)x.z; d[3] = (_Float16)x.w;
        d[4] = (_Float16)y.x; d[5] = (_Float16)y.y;
        d[6] = (_Float16)y.z; d[7] = (_Float16)y.w;
    }
    __syncthreads();

    const int lane = t & 31;
    const int wave = t >> 5;             // 0..7 -> column tile
    const int col0 = wave * 16;
    const int mrow = lane & 15;
    const int half = lane >> 4;
    const v16h* Bf = (const v16h*)Wswz;

    v8f acc = {};
#pragma unroll
    for (int ks = 0; ks < NKS_W; ++ks) {
        v16h a = make_a_frag_h(&sN[mrow * F_IN], ks * 32, lane);
        v16h b = Bf[(wave * NKS_W + ks) * 32 + lane];
        acc = __builtin_amdgcn_wmma_f32_16x16x32_f16(false, a, false, b,
                                                     (short)0, acc, false, false);
    }
    const int n = lane & 15;
    const float bias = Wb[col0 + n];
#pragma unroll
    for (int i = 0; i < 8; ++i) {
        int m = half ? (8 + i) : i;      // N divides by 16 -> no bounds guard
        proj[(row0 + m) * DMODEL + col0 + n] = acc[i] + bias;
    }
}

// Pass 1: per 16-edge tile -- TDM-staged edge tile (f32) -> block f16 convert ->
// WMMA + gather + mish + a-dot, atomic segment-max into nmax.
__global__ __launch_bounds__(256) void k_edge_logits(
        const float* __restrict__ ef, const _Float16* __restrict__ Weswz,
        const float* __restrict__ Web, const float* __restrict__ proj,
        const int* __restrict__ senders, const int* __restrict__ receivers,
        const float* __restrict__ avec, float* __restrict__ logits,
        unsigned* __restrict__ nmax) {
    __shared__ float sF[16 * FE_IN];                    // 4 KB raw f32 (TDM target)
    __shared__ __align__(16) _Float16 sH[16 * FE_IN];   // 2 KB f16 for WMMA
    __shared__ int sS[16], sR[16];

    const int tile = blockIdx.x;
    const int e0 = tile * 16;
    const int t = threadIdx.x;

    if (t < 16) { sS[t] = senders[e0 + t]; sR[t] = receivers[e0 + t]; }
    if (t == 64 && tile + 1 < (int)gridDim.x)
        __builtin_prefetch(&ef[(size_t)(e0 + 16) * FE_IN], 0, 0); // global_prefetch_b8

#if __has_builtin(__builtin_amdgcn_tensor_load_to_lds)
    if (t < 32) {
        // Tensor DMA: 2D tile 64 x 16 of 4-byte elements, row stride 64, into LDS.
        unsigned lds = (unsigned)(size_t)(void*)sF;
        unsigned long long ga = (unsigned long long)(size_t)(ef + (size_t)e0 * FE_IN);
        // D# group0: count=1 | lds_addr | global_addr | type=2 (bits 127:126)
        v4u g0 = { 1u, lds, (unsigned)ga,
                   (unsigned)((ga >> 32) & 0x01FFFFFFu) | 0x80000000u };
        // D# group1: data_size=4B(2)<<16 ; tensor_dim0=64 @bits79:48 ;
        // tensor_dim1=E @bits111:80 ; tile_dim0=64 @bits127:112 ; tile_dim1=16 ;
        // tensor_dim0_stride=64 @bits207:160
        v8i g1 = { (int)(2u << 16),
                   (int)(64u << 16),
                   (int)((unsigned)(E_EDGES & 0xFFFF) << 16),
                   (int)((unsigned)(E_EDGES >> 16) | (64u << 16)),
                   16, 64, 0, 0 };
        v4i gz4 = { 0, 0, 0, 0 };
        v8i gz8 = { 0, 0, 0, 0, 0, 0, 0, 0 };
        __builtin_amdgcn_tensor_load_to_lds(g0, g1, gz4, gz4, gz8, 0);
        __builtin_amdgcn_s_wait_tensorcnt(0);
    }
#else
    for (int i = t; i < 16 * FE_IN; i += 256) sF[i] = ef[(size_t)e0 * FE_IN + i];
#endif
    __syncthreads();

    {   // block-wide f32 -> f16 once (4 elems/thread), instead of 8x per-wave cvt
        const float4* s4 = (const float4*)sF;
        float4 x = s4[t];
        _Float16* d = &sH[t * 4];
        d[0] = (_Float16)x.x; d[1] = (_Float16)x.y;
        d[2] = (_Float16)x.z; d[3] = (_Float16)x.w;
    }
    __syncthreads();

    const int lane = t & 31;
    const int h = t >> 5;                // wave id == head id == column tile
    const int col0 = h * DHEAD;
    const int half = lane >> 4;
    const int mrow = lane & 15;
    const v16h* Bf = (const v16h*)Weswz;

    v8f acc = {};
#pragma unroll
    for (int ks = 0; ks < NKS_WE; ++ks) {
        v16h a = make_a_frag_h(&sH[mrow * FE_IN], ks * 32, lane);
        v16h b = Bf[(h * NKS_WE + ks) * 32 + lane];
        acc = __builtin_amdgcn_wmma_f32_16x16x32_f16(false, a, false, b,
                                                     (short)0, acc, false, false);
    }

    const int n = lane & 15;
    const float an   = avec[h * DHEAD + n];
    const float bias = Web[col0 + n];

#pragma unroll
    for (int i = 0; i < 8; ++i) {
        int m = half ? (8 + i) : i;      // edge-row within tile
        float x = acc[i] + bias
                + proj[sS[m] * DMODEL + col0 + n]
                + proj[sR[m] * DMODEL + col0 + n];
        float term = mishf(x) * an;
#pragma unroll
        for (int d = 8; d >= 1; d >>= 1)         // reduce over n within each 16-lane half
            term += __shfl_xor(term, d, 32);
        if (n == 0) {
            int e = e0 + m;
            logits[e * NHEADS + h] = term;
            atomicMax(&nmax[sR[m] * NHEADS + h], enc_f32(term));
        }
    }
}

// Pass 2: e = exp(logit - max[recv]); segment-sum into nsum; store e in-place.
__global__ void k_expsum(const unsigned* __restrict__ nmax,
                         const int* __restrict__ receivers,
                         float* __restrict__ logits, float* __restrict__ nsum) {
    int idx = blockIdx.x * blockDim.x + threadIdx.x;
    if (idx >= E_EDGES * NHEADS) return;
    int e = idx >> 3, h = idx & 7;
    int r = receivers[e];
    float mx = dec_f32(nmax[r * NHEADS + h]);
    float ev = expf(logits[idx] - mx);
    logits[idx] = ev;
    atomicAdd(&nsum[r * NHEADS + h], ev);
}

// Pass 3: w = e/sum[recv]; out[recv] += w * proj[sender]  (2 edges / 256-thread block)
__global__ __launch_bounds__(256) void k_scatter(const float* __restrict__ logits,
                                                 const float* __restrict__ nsum,
                                                 const float* __restrict__ proj,
                                                 const int* __restrict__ senders,
                                                 const int* __restrict__ receivers,
                                                 float* __restrict__ out) {
    int e = blockIdx.x * 2 + (threadIdx.x >> 7);
    if (e >= E_EDGES) return;
    int d = threadIdx.x & 127;
    int h = d >> 4;
    int r = receivers[e], s = senders[e];
    float w = logits[e * NHEADS + h] / nsum[r * NHEADS + h];
    atomicAdd(&out[r * DMODEL + d], w * proj[s * DMODEL + d]);
}

// ---------- host launcher ----------

extern "C" void kernel_launch(void* const* d_in, const int* in_sizes, int n_in,
                              void* d_out, int out_size, void* d_ws, size_t ws_size,
                              hipStream_t stream) {
    const float* nf   = (const float*)d_in[0];
    const float* ef   = (const float*)d_in[1];
    const int*   snd  = (const int*)d_in[3];
    const int*   rcv  = (const int*)d_in[4];
    const float* Wk   = (const float*)d_in[5];
    const float* Wb   = (const float*)d_in[6];
    const float* Wek  = (const float*)d_in[7];
    const float* Web  = (const float*)d_in[8];
    const float* avec = (const float*)d_in[9];
    float* out = (float*)d_out;

    char* p = (char*)d_ws;
    auto carve = [&](size_t bytes) {
        char* q = p;
        p += (bytes + 255) & ~(size_t)255;
        return q;
    };
    float*     proj   = (float*)carve((size_t)N_NODES * DMODEL * 4);
    float*     logits = (float*)carve((size_t)E_EDGES * NHEADS * 4);
    unsigned*  nmax   = (unsigned*)carve((size_t)N_NODES * NHEADS * 4);
    float*     nsum   = (float*)carve((size_t)N_NODES * NHEADS * 4);
    _Float16*  Wswz   = (_Float16*)carve((size_t)F_IN * DMODEL * 2);
    _Float16*  Weswz  = (_Float16*)carve((size_t)FE_IN * DMODEL * 2);

    {
        int n = N_NODES * DMODEL;
        k_init<<<(n + 255) / 256, 256, 0, stream>>>(out, nsum, nmax);
    }
    {
        int n = F_IN * DMODEL;
        k_convert<<<(n + 255) / 256, 256, 0, stream>>>(Wk, Wek, Wswz, Weswz);
    }
    k_node_proj<<<N_NODES / 16, 256, 0, stream>>>(nf, Wswz, Wb, proj);
    k_edge_logits<<<E_EDGES / 16, 256, 0, stream>>>(ef, Weswz, Web, proj, snd, rcv,
                                                    avec, logits, nmax);
    {
        int n = E_EDGES * NHEADS;
        k_expsum<<<(n + 255) / 256, 256, 0, stream>>>(nmax, rcv, logits, nsum);
    }
    k_scatter<<<E_EDGES / 2, 256, 0, stream>>>(logits, nsum, proj, snd, rcv, out);
}
